// Generator_2078764171343
// MI455X (gfx1250) — matrix operand order repaired and checked by reading, not compiled, not run on hardware
//
#include <hip/hip_runtime.h>

// ---------------------------------------------------------------------------
// Types for CDNA5 WMMA (wave32)
// ---------------------------------------------------------------------------
typedef __bf16 bf16_t;
typedef __attribute__((ext_vector_type(16))) __bf16 v16bf;
typedef __attribute__((ext_vector_type(8)))  __bf16 v8bf;
typedef __attribute__((ext_vector_type(8)))  float  v8f;

#define BATCH   128
#define LAT     512
#define NGATES  2048     // 4*LAT
#define NHID    1024     // 2*LAT
#define NVPAD   32000    // padded vocab head (real = 31999)
#define NVREAL  31999
#define MAXLEN  50
#define END_TOK 1

// ---------------------------------------------------------------------------
// Generic bf16 WMMA GEMM:  out[128,N] = A[128,K] * W[N,K]^T  (+ epilogue)
//   EPI 0: fp32 out = acc + bias[n]
//   EPI 1: bf16 out = relu(acc + bias[n])
//   EPI 2: fp32 out = elu0.1(acc + bias[n])
//   EPI 3: bf16 out = relu(acc*scale[n] + shift[n])        (fused BatchNorm)
// NT = number of 16-wide N tiles per block (4 -> 64 cols, 8 -> 128 cols).
// Block = 256 threads = 8 waves; each wave owns a 16-row M strip.
// B tiles are staged into LDS with GLOBAL_LOAD_ASYNC_TO_LDS_B128 and
// double-buffered: one s_wait_asynccnt + one barrier per k-chunk.
// ---------------------------------------------------------------------------
template<int EPI, int NT>
__global__ __launch_bounds__(256) void gemm_bf16_wmma(
    const bf16_t* __restrict__ A,    // [128, K]
    const bf16_t* __restrict__ W,    // [N, K] row-major
    const float*  __restrict__ p0,   // bias (EPI 0/1/2) or scale (EPI 3)
    const float*  __restrict__ p1,   // shift (EPI 3)
    void* __restrict__ outp,
    int N, int K)
{
    __shared__ __align__(64) bf16_t Bs[2][NT * 16 * 32];  // double-buffered tile

    const int tid  = threadIdx.x;
    const int wave = tid >> 5;
    const int lane = tid & 31;
    const int half = lane >> 4;
    const int l16  = lane & 15;
    const int n0   = blockIdx.x * (NT * 16);
    const int m0   = wave * 16;

    const int ldr  = tid >> 2;        // 0..63 : base B row staged by this thread
    const int lds8 = (tid & 3) * 8;   // 0,8,16,24 : K-element offset (16B)

    // Stage W[n0..n0+NT*16-1, kc..kc+31] into Bs[buf] without touching VGPR
    // data registers: async global->LDS copy, tracked by ASYNCcnt.
    auto stage = [&](int kc, int buf) {
#pragma unroll
        for (int c = 0; c < NT / 4; ++c) {
            const int r = ldr + c * 64;
            unsigned dst = (unsigned)(size_t)&Bs[buf][r * 32 + lds8]; // LDS byte addr
            const bf16_t* src = &W[(size_t)(n0 + r) * K + kc + lds8];
            asm volatile("global_load_async_to_lds_b128 %0, %1, off"
                         :: "v"(dst), "v"(src) : "memory");
        }
    };

    v8f acc[NT] = {};

    stage(0, 0);
    asm volatile("s_wait_asynccnt 0x0" ::: "memory");
    __syncthreads();

    int buf = 0;
    for (int kc = 0; kc < K; kc += 32, buf ^= 1) {
        if (kc + 32 < K) stage(kc + 32, buf ^ 1);   // prefetch next tile (async)

        // ---- A fragment (16x32 bf16 per wave), ISA layout:
        //      VGPR0-3: K = 8*half..8*half+7 ; VGPR4-7: K = 16+8*half..+7
        const size_t arow = (size_t)(m0 + l16) * K + kc;
        v8bf a_lo = *(const v8bf*)&A[arow + half * 8];
        v8bf a_hi = *(const v8bf*)&A[arow + 16 + half * 8];
        v16bf afrag = __builtin_shufflevector(a_lo, a_hi,
            0,1,2,3,4,5,6,7,8,9,10,11,12,13,14,15);

        // ---- NT WMMAs; B fragment from LDS in same K layout as A
#pragma unroll
        for (int t = 0; t < NT; ++t) {
            const bf16_t* brow = &Bs[buf][(t * 16 + l16) * 32];
            v8bf b_lo = *(const v8bf*)&brow[half * 8];
            v8bf b_hi = *(const v8bf*)&brow[16 + half * 8];
            v16bf bfrag = __builtin_shufflevector(b_lo, b_hi,
                0,1,2,3,4,5,6,7,8,9,10,11,12,13,14,15);
            acc[t] = __builtin_amdgcn_wmma_f32_16x16x32_bf16(
                false, afrag, false, bfrag, (short)0, acc[t], false, false);
        }

        // One wait+barrier closes the iteration: proves Bs[buf] fully read by
        // all waves AND Bs[buf^1] fully written by the async copies.
        asm volatile("s_wait_asynccnt 0x0" ::: "memory");
        __syncthreads();
    }

    // ---- epilogue: D layout -> lane column = l16, VGPR r -> row r + 8*half
    float* out_f = (float*)outp;
    bf16_t* out_b = (bf16_t*)outp;
#pragma unroll
    for (int t = 0; t < NT; ++t) {
        const int n = n0 + t * 16 + l16;
        float s0 = p0 ? p0[n] : 0.0f;
        float s1;
        if constexpr (EPI == 3) s1 = p1[n];
#pragma unroll
        for (int r = 0; r < 8; ++r) {
            const int m = m0 + r + 8 * half;
            float v = acc[t][r];
            if constexpr (EPI == 0) {
                out_f[(size_t)m * N + n] = v + s0;
            } else if constexpr (EPI == 1) {
                float x = v + s0;
                out_b[(size_t)m * N + n] = (bf16_t)(x > 0.0f ? x : 0.0f);
            } else if constexpr (EPI == 2) {
                float x = v + s0;
                out_f[(size_t)m * N + n] = x > 0.0f ? x : 0.1f * (__expf(x) - 1.0f);
            } else {
                float x = v * s0 + s1;
                out_b[(size_t)m * N + n] = (bf16_t)(x > 0.0f ? x : 0.0f);
            }
        }
    }
}

// ---------------------------------------------------------------------------
// Setup kernels
// ---------------------------------------------------------------------------
__global__ void k_cast_bf16(const float* __restrict__ s, bf16_t* __restrict__ d, int n) {
    int i = blockIdx.x * blockDim.x + threadIdx.x;
    if (i < n) d[i] = (bf16_t)s[i];
}

// Wcat[n, 0:512] = Wih[n], Wcat[n, 512:1024] = Whh[n]
__global__ void k_build_wcat(const float* __restrict__ Wih, const float* __restrict__ Whh,
                             bf16_t* __restrict__ Wcat) {
    int i = blockIdx.x * blockDim.x + threadIdx.x;      // over 2048*1024
    if (i >= NGATES * NHID) return;
    int n = i / NHID, k = i % NHID;
    float v = (k < LAT) ? Wih[n * LAT + k] : Whh[n * LAT + (k - LAT)];
    Wcat[i] = (bf16_t)v;
}

// Wo2 padded to 32000 rows (pad row = 0), bias padded with -1e30
__global__ void k_build_wo2(const float* __restrict__ Wo2, const float* __restrict__ bo2,
                            bf16_t* __restrict__ Wo2b, float* __restrict__ bo2p) {
    int i = blockIdx.x * blockDim.x + threadIdx.x;      // over 32000*1024
    if (i >= NVPAD * NHID) return;
    int n = i / NHID, k = i % NHID;
    Wo2b[i] = (bf16_t)((n < NVREAL) ? Wo2[(size_t)n * NHID + k] : 0.0f);
    if (k == 0) bo2p[n] = (n < NVREAL) ? bo2[n] : -1e30f;
}

// bias_g = bih + bhh ; fused BN scale/shift for the Wo1 epilogue
__global__ void k_prep(const float* bih, const float* bhh, float* bias_g,
                       const float* bo1, const float* bn_g, const float* bn_b,
                       const float* bn_m, const float* bn_v,
                       float* scale, float* shift) {
    int i = blockIdx.x * blockDim.x + threadIdx.x;
    if (i < NGATES) bias_g[i] = bih[i] + bhh[i];
    if (i < NHID) {
        float sc = bn_g[i] * __frsqrt_rn(bn_v[i] + 1e-5f);
        scale[i] = sc;
        shift[i] = (bo1[i] - bn_m[i]) * sc + bn_b[i];
    }
}

// c = hc[:,512:1024]; xcat[:,512:] = bf16(hc[:,0:512]); xcat[:,0:512] = bf16(emb[0])
__global__ void k_init_state(const float* __restrict__ hc, const float* __restrict__ emb,
                             float* __restrict__ c, bf16_t* __restrict__ xcat) {
    int i = blockIdx.x * blockDim.x + threadIdx.x;      // over 128*512
    if (i >= BATCH * LAT) return;
    int r = i / LAT, j = i % LAT;
    c[i] = hc[r * NHID + LAT + j];
    xcat[r * NHID + LAT + j] = (bf16_t)hc[r * NHID + j];
    xcat[r * NHID + j] = (bf16_t)emb[j];                // START_TOKEN = 0
}

// ---------------------------------------------------------------------------
// LSTM pointwise update (torch gate order i,f,g,o)
// ---------------------------------------------------------------------------
__device__ __forceinline__ float sigm(float x) { return 1.0f / (1.0f + __expf(-x)); }

__global__ void k_lstm_update(const float* __restrict__ gates, float* __restrict__ c,
                              bf16_t* __restrict__ xcat, bf16_t* __restrict__ hrelu) {
    int i = blockIdx.x * blockDim.x + threadIdx.x;      // over 128*512
    if (i >= BATCH * LAT) return;
    int r = i / LAT, j = i % LAT;
    const float* g = gates + (size_t)r * NGATES;
    float gi = g[j], gf = g[LAT + j], gg = g[2 * LAT + j], go = g[3 * LAT + j];
    float cn = sigm(gf) * c[i] + sigm(gi) * tanhf(gg);
    float h  = sigm(go) * tanhf(cn);
    c[i] = cn;
    xcat[r * NHID + LAT + j] = (bf16_t)h;
    hrelu[i] = (bf16_t)(h > 0.0f ? h : 0.0f);
}

// ---------------------------------------------------------------------------
// Sampling / softmax reductions over the logits row (32000 incl. pad)
// ---------------------------------------------------------------------------
__device__ __forceinline__ float gumbel_noise(unsigned step, unsigned r, unsigned v) {
    unsigned x = step * 0x9E3779B9u ^ (r * 0x85EBCA6Bu + 0x27D4EB2Du) ^ (v * 0xC2B2AE35u);
    x ^= x >> 16; x *= 0x7FEB352Du; x ^= x >> 15; x *= 0x846CA68Bu; x ^= x >> 16;
    float u = ((x >> 8) + 0.5f) * (1.0f / 16777216.0f);
    return -__logf(-__logf(u));
}

__global__ void k_pass1(const float* __restrict__ logits, float* __restrict__ rowmax,
                        int* __restrict__ samples, int step) {
    __shared__ float sm[256], sg[256];
    __shared__ int   si[256];
    int r = blockIdx.x, tid = threadIdx.x;
    const float* L = logits + (size_t)r * NVPAD;
    float m = -3.0e38f, gm = -3.0e38f; int gi = 0;
    for (int v = tid; v < NVPAD; v += 256) {
        float l = L[v];
        m = fmaxf(m, l);
        float g = l + gumbel_noise((unsigned)step, (unsigned)r, (unsigned)v);
        if (g > gm) { gm = g; gi = v; }
    }
    sm[tid] = m; sg[tid] = gm; si[tid] = gi;
    __syncthreads();
    for (int s = 128; s > 0; s >>= 1) {
        if (tid < s) {
            sm[tid] = fmaxf(sm[tid], sm[tid + s]);
            if (sg[tid + s] > sg[tid]) { sg[tid] = sg[tid + s]; si[tid] = si[tid + s]; }
        }
        __syncthreads();
    }
    if (tid == 0) { rowmax[r] = sm[0]; samples[step * BATCH + r] = si[0]; }
}

__global__ void k_pass2(const float* __restrict__ logits, const float* __restrict__ rowmax,
                        const int* __restrict__ samples, const float* __restrict__ emb,
                        bf16_t* __restrict__ xcat, float* __restrict__ lps,
                        float* __restrict__ ents, int step) {
    __shared__ float ss[256], st[256];
    int r = blockIdx.x, tid = threadIdx.x;
    const float* L = logits + (size_t)r * NVPAD;
    float m = rowmax[r];
    float S = 0.0f, T = 0.0f;
    for (int v = tid; v < NVPAD; v += 256) {
        float l = L[v];
        float e = __expf(l - m);
        S += e; T += e * l;
    }
    ss[tid] = S; st[tid] = T;
    __syncthreads();
    for (int s = 128; s > 0; s >>= 1) {
        if (tid < s) { ss[tid] += ss[tid + s]; st[tid] += st[tid + s]; }
        __syncthreads();
    }
    int tok = samples[step * BATCH + r];
    if (tid == 0) {
        float logS = __logf(ss[0]);
        lps[step * BATCH + r]  = L[tok] - m - logS;           // log-prob of sample
        ents[step * BATCH + r] = m + logS - st[0] / ss[0];    // entropy
    }
    // embedding gather for next step's input
    for (int j = tid; j < LAT; j += 256)
        xcat[r * NHID + j] = (bf16_t)emb[(size_t)tok * LAT + j];
}

// ---------------------------------------------------------------------------
// Output assembly: replicates first-END masking incl. the length-1 quirk
// ---------------------------------------------------------------------------
__global__ void k_finalize(const int* __restrict__ samples, const float* __restrict__ lps,
                           const float* __restrict__ ents, float* __restrict__ out) {
    int r = threadIdx.x;                 // one block of 128 threads
    if (r >= BATCH) return;
    int end_pos = 0; bool found = false;
    for (int s = 0; s < MAXLEN; ++s) {
        if (!found && samples[s * BATCH + r] == END_TOK) { end_pos = s; found = true; }
    }
    int seq_len = end_pos + 1;
    if (seq_len == 1) seq_len = MAXLEN;  // torch quirk (also covers no-END case)
    float* out_x  = out;
    float* out_lp = out + BATCH * MAXLEN;
    float* out_e  = out + 2 * BATCH * MAXLEN;
    float esum = 0.0f;
    for (int s = 0; s < MAXLEN; ++s) {
        bool msk = s < seq_len;
        out_x[r * MAXLEN + s]  = msk ? (float)(samples[s * BATCH + r] + 1) : 0.0f;
        out_lp[r * MAXLEN + s] = msk ? lps[s * BATCH + r] : 0.0f;
        if (msk) esum += ents[s * BATCH + r];
    }
    out_e[r] = esum / (float)seq_len;
}

// ---------------------------------------------------------------------------
// Host driver (graph-capturable: all launches on `stream`, ws-only scratch)
// ---------------------------------------------------------------------------
extern "C" void kernel_launch(void* const* d_in, const int* in_sizes, int n_in,
                              void* d_out, int out_size, void* d_ws, size_t ws_size,
                              hipStream_t stream) {
    const float* z    = (const float*)d_in[0];
    const float* emb  = (const float*)d_in[1];
    const float* Wp1  = (const float*)d_in[2];
    const float* bp1  = (const float*)d_in[3];
    const float* Wp2  = (const float*)d_in[4];
    const float* bp2  = (const float*)d_in[5];
    const float* Wih  = (const float*)d_in[6];
    const float* Whh  = (const float*)d_in[7];
    const float* bih  = (const float*)d_in[8];
    const float* bhh  = (const float*)d_in[9];
    const float* Wo1  = (const float*)d_in[10];
    const float* bo1  = (const float*)d_in[11];
    const float* bn_g = (const float*)d_in[12];
    const float* bn_b = (const float*)d_in[13];
    const float* bn_m = (const float*)d_in[14];
    const float* bn_v = (const float*)d_in[15];
    const float* Wo2  = (const float*)d_in[16];
    const float* bo2  = (const float*)d_in[17];
    // d_in[18] = max_len (known constant 50)

    char* base = (char*)d_ws;
    size_t off = 0;
    auto alloc = [&](size_t bytes) -> void* {
        void* p = base + off;
        off = (off + bytes + 255) & ~(size_t)255;
        return p;
    };
    bf16_t* Wo2b  = (bf16_t*)alloc((size_t)NVPAD * NHID * 2);   // 65.5 MB (L2-resident)
    bf16_t* Wcatb = (bf16_t*)alloc((size_t)NGATES * NHID * 2);
    bf16_t* Wp1b  = (bf16_t*)alloc((size_t)NHID * LAT * 2);
    bf16_t* Wp2b  = (bf16_t*)alloc((size_t)NHID * NHID * 2);
    bf16_t* Wo1b  = (bf16_t*)alloc((size_t)NHID * LAT * 2);
    float*  biasg = (float*)alloc(NGATES * 4);
    float*  bnsc  = (float*)alloc(NHID * 4);
    float*  bnsh  = (float*)alloc(NHID * 4);
    float*  bo2p  = (float*)alloc(NVPAD * 4);
    bf16_t* zb    = (bf16_t*)alloc((size_t)BATCH * LAT * 2);
    bf16_t* h1b   = (bf16_t*)alloc((size_t)BATCH * NHID * 2);
    float*  hc    = (float*)alloc((size_t)BATCH * NHID * 4);
    float*  cst   = (float*)alloc((size_t)BATCH * LAT * 4);
    bf16_t* xcat  = (bf16_t*)alloc((size_t)BATCH * NHID * 2);
    bf16_t* hrelu = (bf16_t*)alloc((size_t)BATCH * LAT * 2);
    bf16_t* actb  = (bf16_t*)alloc((size_t)BATCH * NHID * 2);
    float*  gates = (float*)alloc((size_t)BATCH * NGATES * 4);
    float*  logit = (float*)alloc((size_t)BATCH * NVPAD * 4);  // 16.4 MB
    float*  rowmx = (float*)alloc(BATCH * 4);
    int*    smpl  = (int*)alloc((size_t)MAXLEN * BATCH * 4);
    float*  lps   = (float*)alloc((size_t)MAXLEN * BATCH * 4);
    float*  ents  = (float*)alloc((size_t)MAXLEN * BATCH * 4);
    (void)ws_size; (void)n_in; (void)in_sizes; (void)out_size;

    const dim3 T256(256);
    // ---- one-time weight conversion / folding
    k_cast_bf16<<<(NHID * LAT + 255) / 256, T256, 0, stream>>>(Wp1, Wp1b, NHID * LAT);
    k_cast_bf16<<<(NHID * NHID + 255) / 256, T256, 0, stream>>>(Wp2, Wp2b, NHID * NHID);
    k_cast_bf16<<<(NHID * LAT + 255) / 256, T256, 0, stream>>>(Wo1, Wo1b, NHID * LAT);
    k_cast_bf16<<<(BATCH * LAT + 255) / 256, T256, 0, stream>>>(z, zb, BATCH * LAT);
    k_build_wcat<<<(NGATES * NHID + 255) / 256, T256, 0, stream>>>(Wih, Whh, Wcatb);
    k_build_wo2<<<((size_t)NVPAD * NHID + 255) / 256, T256, 0, stream>>>(Wo2, bo2, Wo2b, bo2p);
    k_prep<<<(NGATES + 255) / 256, T256, 0, stream>>>(bih, bhh, biasg, bo1,
                                                      bn_g, bn_b, bn_m, bn_v, bnsc, bnsh);

    // ---- projection: z -> (h, c)
    gemm_bf16_wmma<1, 4><<<NHID / 64, T256, 0, stream>>>(zb, Wp1b, bp1, nullptr,
                                                         h1b, NHID, LAT);
    gemm_bf16_wmma<2, 4><<<NHID / 64, T256, 0, stream>>>(h1b, Wp2b, bp2, nullptr,
                                                         hc, NHID, NHID);
    k_init_state<<<(BATCH * LAT + 255) / 256, T256, 0, stream>>>(hc, emb, cst, xcat);

    // ---- 50 autoregressive steps
    for (int s = 0; s < MAXLEN; ++s) {
        gemm_bf16_wmma<0, 4><<<NGATES / 64, T256, 0, stream>>>(xcat, Wcatb, biasg, nullptr,
                                                               gates, NGATES, NHID);
        k_lstm_update<<<(BATCH * LAT + 255) / 256, T256, 0, stream>>>(gates, cst, xcat, hrelu);
        gemm_bf16_wmma<3, 4><<<NHID / 64, T256, 0, stream>>>(hrelu, Wo1b, bnsc, bnsh,
                                                             actb, NHID, LAT);
        gemm_bf16_wmma<0, 8><<<NVPAD / 128, T256, 0, stream>>>(actb, Wo2b, bo2p, nullptr,
                                                               logit, NVPAD, NHID);
        k_pass1<<<BATCH, T256, 0, stream>>>(logit, rowmx, smpl, s);
        k_pass2<<<BATCH, T256, 0, stream>>>(logit, rowmx, smpl, emb, xcat, lps, ents, s);
    }

    k_finalize<<<1, dim3(128), 0, stream>>>(smpl, lps, ents, (float*)d_out);
}
